// GPPM_79594333929561
// MI455X (gfx1250) — matrix operand order repaired and checked by reading, not compiled, not run on hardware
//
#include <hip/hip_runtime.h>
#include <hip/hip_bf16.h>
#include <math.h>

typedef __attribute__((ext_vector_type(16))) _Float16 v16h;
typedef __attribute__((ext_vector_type(8)))  float    v8f;

#define NFEAT 256
#define HID   32
#define NCLS  64
#define ALPHA 0.9f
#define BETA  0.1f
#define PROP  4

// ---------------------------------------------------------------------------
// WMMA fragment loaders (layouts per CDNA5 ISA 7.12.2, wave32)
// ---------------------------------------------------------------------------

// A-matrix 16x32 f16 fragment from global f32, row-major, leading dim ld.
// lanes 0-15: M=lane, K-halves {0..7, 16..23}; lanes 16-31: M=lane-16,
// K-halves {8..15, 24..31}.
__device__ __forceinline__ v16h load_a_frag_g(const float* __restrict__ xk,
                                              int ld, int row0, int nrows,
                                              int lane) {
  int m = lane & 15;
  int row = row0 + m;
  if (row >= nrows) row = nrows - 1;       // clamp for partial tail tile
  const float* r = xk + (size_t)row * ld;
  int kh = (lane >> 4) * 8;
  v16h a;
#pragma unroll
  for (int j = 0; j < 4; ++j) {
    int k0 = kh + 2 * j;
    a[2 * j]     = (_Float16)r[k0];
    a[2 * j + 1] = (_Float16)r[k0 + 1];
  }
#pragma unroll
  for (int j = 0; j < 4; ++j) {
    int k0 = 16 + kh + 2 * j;
    a[8 + 2 * j]     = (_Float16)r[k0];
    a[8 + 2 * j + 1] = (_Float16)r[k0 + 1];
  }
  return a;
}

// A fragment from LDS f16 tile (16 rows x 32 cols, row-major).
__device__ __forceinline__ v16h load_a_frag_lds(const _Float16* __restrict__ h,
                                                int lane) {
  int m = lane & 15;
  const _Float16* r = h + m * 32;
  int kh = (lane >> 4) * 8;
  v16h a;
#pragma unroll
  for (int j = 0; j < 4; ++j) {
    int k0 = kh + 2 * j;
    a[2 * j]     = r[k0];
    a[2 * j + 1] = r[k0 + 1];
  }
#pragma unroll
  for (int j = 0; j < 4; ++j) {
    int k0 = 16 + kh + 2 * j;
    a[8 + 2 * j]     = r[k0];
    a[8 + 2 * j + 1] = r[k0 + 1];
  }
  return a;
}

// B-matrix 32x16 f16 fragment from global f32 W[K][ld], columns ncol0..+15.
// lanes 0-15: N=ncol0+lane, K={0..7, 16..23}; lanes 16-31: K={8..15, 24..31}.
__device__ __forceinline__ v16h load_b_frag(const float* __restrict__ Wk,
                                            int ld, int ncol0, int lane) {
  int n  = ncol0 + (lane & 15);
  int kb = (lane >> 4) * 8;
  v16h b;
#pragma unroll
  for (int j = 0; j < 8; ++j) {
    b[2 * j]     = (_Float16)Wk[(size_t)(kb + 2 * j + (j >= 4 ? 8 : 0)) * ld + n];
    b[2 * j + 1] = (_Float16)Wk[(size_t)(kb + 2 * j + 1 + (j >= 4 ? 8 : 0)) * ld + n];
  }
  return b;
}

__device__ __forceinline__ v8f wmma_f16(v16h a, v16h b, v8f c) {
  return __builtin_amdgcn_wmma_f32_16x16x32_f16(false, a, false, b,
                                                (short)0, c, false, false);
}

// ---------------------------------------------------------------------------
// Kernel 1: p = softmax(relu(x@W1+b1)@W2+b2), written to both P and Q.
// One wave per 16-node tile, 8 waves (128 nodes) per block.
// ---------------------------------------------------------------------------
__global__ void __launch_bounds__(256)
mlp_softmax_wmma(const float* __restrict__ x, const float* __restrict__ W1,
                 const float* __restrict__ b1, const float* __restrict__ W2,
                 const float* __restrict__ b2, float* __restrict__ P,
                 float* __restrict__ Q, int nNodes) {
  __shared__ _Float16 hbuf[8][16 * HID];   // 8 KB
  __shared__ float    lbuf[8][16 * NCLS];  // 32 KB
  __shared__ float    smax[8][16];
  __shared__ float    sinv[8][16];

  int lane = threadIdx.x & 31;
  int wv   = threadIdx.x >> 5;
  int row0 = (blockIdx.x * 8 + wv) * 16;

  // ---- GEMM1: [16x256] x [256x32], K in 8 steps of 32 ----
  v8f acc0 = {};
  v8f acc1 = {};
#pragma unroll
  for (int ks = 0; ks < NFEAT; ks += 32) {
    v16h a   = load_a_frag_g(x + ks, NFEAT, row0, nNodes, lane);
    v16h bA  = load_b_frag(W1 + (size_t)ks * HID, HID, 0, lane);
    v16h bB  = load_b_frag(W1 + (size_t)ks * HID, HID, 16, lane);
    acc0 = wmma_f16(a, bA, acc0);
    acc1 = wmma_f16(a, bB, acc1);
  }

  // ---- bias + relu, stage h to LDS (D layout: row = r + 8*(lane>>4)) ----
  int rbase = 8 * (lane >> 4);
  int c0    = lane & 15;
#pragma unroll
  for (int r = 0; r < 8; ++r) {
    float h0 = fmaxf(acc0[r] + b1[c0], 0.0f);
    float h1 = fmaxf(acc1[r] + b1[c0 + 16], 0.0f);
    hbuf[wv][(rbase + r) * HID + c0]      = (_Float16)h0;
    hbuf[wv][(rbase + r) * HID + c0 + 16] = (_Float16)h1;
  }
  __syncthreads();

  // ---- GEMM2: [16x32] x [32x64], single K step, 4 N tiles ----
  v16h ah = load_a_frag_lds(hbuf[wv], lane);
  v8f l0 = {}, l1 = {}, l2 = {}, l3 = {};
  l0 = wmma_f16(ah, load_b_frag(W2, NCLS, 0, lane), l0);
  l1 = wmma_f16(ah, load_b_frag(W2, NCLS, 16, lane), l1);
  l2 = wmma_f16(ah, load_b_frag(W2, NCLS, 32, lane), l2);
  l3 = wmma_f16(ah, load_b_frag(W2, NCLS, 48, lane), l3);

#pragma unroll
  for (int r = 0; r < 8; ++r) {
    lbuf[wv][(rbase + r) * NCLS + c0]      = l0[r] + b2[c0];
    lbuf[wv][(rbase + r) * NCLS + c0 + 16] = l1[r] + b2[c0 + 16];
    lbuf[wv][(rbase + r) * NCLS + c0 + 32] = l2[r] + b2[c0 + 32];
    lbuf[wv][(rbase + r) * NCLS + c0 + 48] = l3[r] + b2[c0 + 48];
  }
  __syncthreads();

  // ---- row softmax stats (lanes 0..15, one row each) ----
  if (lane < 16) {
    const float* lr = &lbuf[wv][lane * NCLS];
    float mx = lr[0];
#pragma unroll
    for (int c = 1; c < NCLS; ++c) mx = fmaxf(mx, lr[c]);
    float s = 0.0f;
#pragma unroll
    for (int c = 0; c < NCLS; ++c) s += __expf(lr[c] - mx);
    smax[wv][lane] = mx;
    sinv[wv][lane] = 1.0f / s;
  }
  __syncthreads();

  // ---- normalize and write p into P and Q (Q = self-loop init) ----
  for (int r = 0; r < 16; ++r) {
    int node = row0 + r;
    if (node < nNodes) {
      float mx = smax[wv][r];
      float iv = sinv[wv][r];
      size_t base = (size_t)node * NCLS;
#pragma unroll
      for (int c = lane; c < NCLS; c += 32) {
        float v = __expf(lbuf[wv][r * NCLS + c] - mx) * iv;
        P[base + c] = v;
        Q[base + c] = v;
      }
    }
  }
}

// ---------------------------------------------------------------------------
// Kernel 2: Q[row[e]] += P[col[e]]  (16 threads per edge, float4 groups).
// P/Q are L2-resident (25.6 MB each vs 192 MB L2) -> atomics serviced in L2.
// ---------------------------------------------------------------------------
__global__ void __launch_bounds__(256)
scatter_edges(const int* __restrict__ erow, const int* __restrict__ ecol,
              const float* __restrict__ P, float* __restrict__ Q, int nEdges) {
  unsigned long long gid =
      (unsigned long long)blockIdx.x * blockDim.x + threadIdx.x;
  int e = (int)(gid >> 4);
  if (e >= nEdges) return;
  int g   = ((int)gid & 15) * 4;
  int src = ecol[e];
  int dst = erow[e];
  const float4 v = *(const float4*)(P + (size_t)src * NCLS + g);
  float* q = Q + (size_t)dst * NCLS + g;
  atomicAdd(q + 0, v.x);
  atomicAdd(q + 1, v.y);
  atomicAdd(q + 2, v.z);
  atomicAdd(q + 3, v.w);
}

// ---------------------------------------------------------------------------
// Kernel 3: new = ALPHA*Q + BETA*P; P = Q = new; out (+)= softmax(new).
// One wave per node: lane handles cols lane and lane+32.
// ---------------------------------------------------------------------------
__global__ void __launch_bounds__(256)
combine_softmax(float* __restrict__ P, float* __restrict__ Q,
                float* __restrict__ out, int nNodes, int first) {
  int wv   = threadIdx.x >> 5;
  int lane = threadIdx.x & 31;
  int node = blockIdx.x * 8 + wv;
  if (node >= nNodes) return;
  size_t base = (size_t)node * NCLS;

  float v0 = ALPHA * Q[base + lane]      + BETA * P[base + lane];
  float v1 = ALPHA * Q[base + lane + 32] + BETA * P[base + lane + 32];

  // carry state to next step: P = new; Q pre-initialized with self-loop term
  P[base + lane]      = v0;
  P[base + lane + 32] = v1;
  Q[base + lane]      = v0;
  Q[base + lane + 32] = v1;

  float mx = fmaxf(v0, v1);
#pragma unroll
  for (int off = 16; off > 0; off >>= 1) mx = fmaxf(mx, __shfl_xor(mx, off, 32));
  float e0 = __expf(v0 - mx);
  float e1 = __expf(v1 - mx);
  float s  = e0 + e1;
#pragma unroll
  for (int off = 16; off > 0; off >>= 1) s += __shfl_xor(s, off, 32);
  float iv = 1.0f / s;

  if (first) {
    out[base + lane]      = e0 * iv;
    out[base + lane + 32] = e1 * iv;
  } else {
    out[base + lane]      += e0 * iv;
    out[base + lane + 32] += e1 * iv;
  }
}

// ---------------------------------------------------------------------------
extern "C" void kernel_launch(void* const* d_in, const int* in_sizes, int n_in,
                              void* d_out, int out_size, void* d_ws,
                              size_t ws_size, hipStream_t stream) {
  const float* x    = (const float*)d_in[0];
  const int*   eIdx = (const int*)d_in[1];
  const float* W1   = (const float*)d_in[2];
  const float* b1   = (const float*)d_in[3];
  const float* W2   = (const float*)d_in[4];
  const float* b2   = (const float*)d_in[5];
  float*       out  = (float*)d_out;

  int nNodes = in_sizes[0] / NFEAT;
  int nEdges = in_sizes[1] / 2;

  float* P = (float*)d_ws;                       // [nNodes][64]
  float* Q = P + (size_t)nNodes * NCLS;          // [nNodes][64]

  const int* erow = eIdx;            // edgeIndex[0]
  const int* ecol = eIdx + nEdges;   // edgeIndex[1]

  // 1) pseudo-labels via WMMA MLP; writes P and Q (= p)
  int blocks1 = (nNodes + 127) / 128;
  mlp_softmax_wmma<<<blocks1, 256, 0, stream>>>(x, W1, b1, W2, b2, P, Q,
                                                nNodes);

  // 2) 4 propagation steps
  unsigned long long tE = (unsigned long long)nEdges * 16ull;
  int blocksE = (int)((tE + 255ull) / 256ull);
  int blocksC = (nNodes + 7) / 8;
  for (int s = 0; s < PROP; ++s) {
    scatter_edges<<<blocksE, 256, 0, stream>>>(erow, ecol, P, Q, nEdges);
    combine_softmax<<<blocksC, 256, 0, stream>>>(P, Q, out, nNodes, s == 0);
  }
}
